// DCRNN_57157424775354
// MI455X (gfx1250) — compile-verified, matrix operand order checked by reading, not run on hardware
//
#include <hip/hip_runtime.h>
#include <hip/hip_bf16.h>

typedef __attribute__((ext_vector_type(16))) __bf16 v16bf;
typedef __attribute__((ext_vector_type(8)))  float  v8f;
typedef __attribute__((ext_vector_type(4)))  unsigned int u32x4;
typedef __attribute__((ext_vector_type(8)))  int    i32x8;
typedef __attribute__((ext_vector_type(4)))  int    i32x4;

#define NNODE   18
#define DIMX    32
#define HID     64
#define TSTEPS  128
#define BTOT    512

#define BB      2                 // batches per block
#define ROWS    (BB*NNODE)        // 36
#define PADR    48                // padded rows (3 tiles of 16)
#define MT      3                 // row tiles
#define NWAVES  8
#define NTHREADS 256
#define MAXACC  3                 // max tiles per wave (gate: 24/8 = 3)

#define KT0     15                // 480/32 k-tiles (layer0)
#define KT1     20                // 640/32 k-tiles (layer1)
#define NT_G    8                 // 128/16 col tiles (gates)
#define NT_C    4                 // 64/16  col tiles (cand)

#define BSTAGE_ELEMS (4*NT_G*512) // max chunk: 4 k-tiles x 8 col tiles = 16384 bf16 (32KB)

// weight-fragment offsets in d_ws (bf16 elements); each tile = 32 lanes * 16 vals
#define OFF_WG0 0
#define OFF_WC0 (KT0*NT_G*512)                 // 61440
#define OFF_WG1 (OFF_WC0 + KT0*NT_C*512)       // 92160
#define OFF_WC1 (OFF_WG1 + KT1*NT_G*512)       // 174080

// ---------------------------------------------------------------------------
// Pre-swizzle f32 weights [K,OUT] row-major -> per-lane WMMA B-fragment layout:
// dst[((kt*NT + nt)*32 + lane)*16 + j]  holds W[k, n] with
//   n = nt*16 + (lane&15)
//   k = kt*32 + (j&7) + (j>>3)*16 + ((lane>>4)&1)*8
// ---------------------------------------------------------------------------
__global__ void dcrnn_prep_weights(const float* __restrict__ w,
                                   __bf16* __restrict__ dst, int K, int OUT) {
  int NTt   = OUT >> 4;
  int total = (K >> 5) * NTt * 512;
  for (int i = blockIdx.x * blockDim.x + threadIdx.x; i < total;
       i += gridDim.x * blockDim.x) {
    int j    = i & 15;
    int lane = (i >> 4) & 31;
    int tile = i >> 9;
    int kt   = tile / NTt;
    int nt   = tile - kt * NTt;
    int n    = nt * 16 + (lane & 15);
    int koff = (j & 7) + ((j >> 3) << 4) + (((lane >> 4) & 1) << 3);
    int k    = kt * 32 + koff;
    dst[i]   = (__bf16)w[(size_t)k * OUT + n];
  }
}

// ---------------------------------------------------------------------------
// TDM: DMA a contiguous run of len_elems bf16 from global into LDS.
// D# per CDNA5 ISA §8: group0 = {count|flags, lds_addr, global_addr lo,
// global_addr hi | type=2}; group1 = {data_size=2B, tensor_dim0=len,
// tensor_dim1=1, tile_dim0=len (<=16384), stride0=len}; groups 2/3 zero.
// Tracked with TENSORcnt.  (6-arg builtin form: extra zero int32x8 group.)
// ---------------------------------------------------------------------------
__device__ __forceinline__ void tdm_load_weights(const __bf16* gsrc,
                                                 __bf16* lds_dst, int len) {
  unsigned long long ga = (unsigned long long)(size_t)gsrc;
  u32x4 g0;
  g0[0] = 1u;                                            // count=1, user mode
  g0[1] = (unsigned)(unsigned long long)(size_t)lds_dst; // LDS byte offset (low 32b)
  g0[2] = (unsigned)(ga & 0xffffffffu);                  // global_addr[31:0]
  g0[3] = (unsigned)((ga >> 32) & 0x01ffffffu) | (2u << 30); // [56:32] | type=2
  i32x8 g1;
  g1[0] = 1 << 16;                                       // data_size = 2 bytes
  g1[1] = (len & 0xffff) << 16;                          // tensor_dim0[15:0]
  g1[2] = ((len >> 16) & 0xffff) | (1 << 16);            // tensor_dim0[31:16] | tensor_dim1=1
  g1[3] = (len & 0xffff) << 16;                          // tile_dim0 (<= 16384)
  g1[4] = 0;                                             // tile_dim1/2 unused
  g1[5] = len;                                           // tensor_dim0_stride lo
  g1[6] = 0;
  g1[7] = 0;
  i32x4 z4 = {0, 0, 0, 0};
  i32x8 z8 = {0, 0, 0, 0, 0, 0, 0, 0};
  __builtin_amdgcn_tensor_load_to_lds(g0, g1, z4, z4, z8, 0);
}

// stage one f32 value into the A-fragment-layout LDS buffer (16-bit A 16x32 layout)
__device__ __forceinline__ void stage_frag(__bf16* frag, int row, int c, float v) {
  int kc   = c >> 5;
  int ko   = c & 31;
  int mt   = row >> 4;
  int ml   = row & 15;
  int lane = ml + (((ko >> 3) & 1) << 4);
  int j    = (ko & 7) + (((ko >> 4) & 1) << 3);
  frag[(((kc * MT + mt) << 5) + lane) * 16 + j] = (__bf16)v;
}

// chunk 0: concatenated input  (MODE 0:[x|ha] 1:[x|r*ha] 2:[ha|hb] 3:[ha|r*hb])
template<int MODE, int C>
__device__ __forceinline__ void build_cat(const float* xg,
    const float* sHa, const float* sHb, const float* sR,
    float* sCat, __bf16* frag) {
  for (int idx = threadIdx.x; idx < PADR * C; idx += NTHREADS) {
    int row = idx / C;
    int c   = idx - row * C;
    float v = 0.f;
    if (row < ROWS) {
      if (MODE <= 1) {
        if (c < DIMX) {
          int bl = row / NNODE, n = row - bl * NNODE;
          v = xg[(size_t)bl * (TSTEPS * NNODE * DIMX) + n * DIMX + c];
        } else {
          int cc = c - DIMX;
          float h = sHa[row * HID + cc];
          v = (MODE == 0) ? h : sR[row * HID + cc] * h;
        }
      } else {
        if (c < HID) {
          v = sHa[row * HID + c];
        } else {
          int cc = c - HID;
          float h = sHb[row * HID + cc];
          v = (MODE == 2) ? h : sR[row * HID + cc] * h;
        }
      }
      sCat[row * 128 + c] = v;
    }
    stage_frag(frag, row, c, v);
  }
}

// diffusion chunk: dst = alpha * (S[sIdx] @ src) + beta * x0   (per-batch 18x18 mix)
template<int C>
__device__ __forceinline__ void build_diff(const float* sS, int sIdx,
    const float* src, const float* x0, float alpha, float beta,
    float* dst, __bf16* frag) {
  for (int idx = threadIdx.x; idx < PADR * C; idx += NTHREADS) {
    int row = idx / C;
    int c   = idx - row * C;
    float v = 0.f;
    if (row < ROWS) {
      int bl = row / NNODE, n = row - bl * NNODE;
      const float* Srow = sS + (sIdx * NNODE + n) * NNODE;
      const float* srow = src + bl * NNODE * 128 + c;
      float acc = 0.f;
#pragma unroll
      for (int m = 0; m < NNODE; ++m) acc += Srow[m] * srow[m * 128];
      v = alpha * acc;
      if (x0) v += beta * x0[row * 128 + c];
      dst[row * 128 + c] = v;
    }
    stage_frag(frag, row, c, v);
  }
}

// accumulate one staged feature chunk into per-wave WMMA accumulators.
// A fragments from sFrag, B fragments from the TDM-staged LDS buffer.
template<int C, int NT>
__device__ __forceinline__ void wmma_accum(const __bf16* sBW,
    const __bf16* frag, v8f* acc, int wave, int lane) {
  constexpr int kcN   = C >> 5;
  constexpr int TOTAL = MT * NT;
  constexpr int TPW   = (TOTAL + NWAVES - 1) / NWAVES;
#pragma unroll
  for (int kc = 0; kc < kcN; ++kc) {
#pragma unroll
    for (int a = 0; a < TPW; ++a) {
      int ti = wave * TPW + a;
      if (TOTAL % NWAVES == 0 || ti < TOTAL) {
        int mt = ti / NT;
        int nt = ti - mt * NT;
        v16bf av = *(const v16bf*)(frag + (((kc * MT + mt) << 5) + lane) * 16);
        v16bf bv = *(const v16bf*)(sBW  + (((kc * NT + nt) << 5) + lane) * 16);
        acc[a] = __builtin_amdgcn_wmma_f32_16x16x32_bf16(
            false, av, false, bv, (short)0, acc[a], false, false);
      }
    }
  }
}

// full diffusion-conv: 5 Chebyshev chunks; per chunk the TDM DMAs the B
// fragments into LDS while all waves build/stage the A-side features.
template<int MODE, int C, int NT>
__device__ __forceinline__ void dconv(const float* xg,
    const float* sS, const float* sHa, const float* sHb, const float* sR,
    float* sCat, float* sB1, float* sB2, __bf16* frag, __bf16* sBW,
    const __bf16* wbase, v8f* acc, int wave, int lane) {
  constexpr int kcN   = C >> 5;
  constexpr int CHUNK = kcN * NT * 512;   // bf16 elems per feature chunk
  constexpr int TPW   = (MT * NT + NWAVES - 1) / NWAVES;
#pragma unroll
  for (int a = 0; a < TPW; ++a)
#pragma unroll
    for (int r = 0; r < 8; ++r) acc[a][r] = 0.f;

  if (wave == 0) tdm_load_weights(wbase, sBW, CHUNK);
  build_cat<MODE, C>(xg, sHa, sHb, sR, sCat, frag);
  if (wave == 0) __builtin_amdgcn_s_wait_tensorcnt(0);
  __syncthreads();
  wmma_accum<C, NT>(sBW, frag, acc, wave, lane);
  __syncthreads();
#pragma unroll
  for (int s = 0; s < 2; ++s) {
    if (wave == 0) tdm_load_weights(wbase + (size_t)(1 + 2 * s) * CHUNK, sBW, CHUNK);
    build_diff<C>(sS, s, sCat, nullptr, 1.f, 0.f, sB1, frag);   // x1 = S x0
    if (wave == 0) __builtin_amdgcn_s_wait_tensorcnt(0);
    __syncthreads();
    wmma_accum<C, NT>(sBW, frag, acc, wave, lane);
    __syncthreads();
    if (wave == 0) tdm_load_weights(wbase + (size_t)(2 + 2 * s) * CHUNK, sBW, CHUNK);
    build_diff<C>(sS, s, sB1, sCat, 2.f, -1.f, sB2, frag);      // x2 = 2 S x1 - x0
    if (wave == 0) __builtin_amdgcn_s_wait_tensorcnt(0);
    __syncthreads();
    wmma_accum<C, NT>(sBW, frag, acc, wave, lane);
    __syncthreads();
  }
}

// gate epilogue: sigmoid(acc + b) -> r,u  (C/D fragment layout -> LDS)
__device__ __forceinline__ void write_gate(const v8f* acc, const float* bg,
    float* sR, float* sU, int wave, int lane) {
  constexpr int TPW = (MT * NT_G + NWAVES - 1) / NWAVES;  // 3
#pragma unroll
  for (int a = 0; a < TPW; ++a) {
    int ti = wave * TPW + a;
    int mt = ti / NT_G, nt = ti - mt * NT_G;
    int col   = nt * 16 + (lane & 15);
    int rbase = mt * 16 + ((lane >> 4) << 3);
#pragma unroll
    for (int r = 0; r < 8; ++r) {
      int row = rbase + r;
      if (row < ROWS) {
        float v = acc[a][r] + bg[col];
        float s = 1.f / (1.f + __expf(-v));
        if (col < HID) sR[row * HID + col] = s;
        else           sU[row * HID + col - HID] = s;
      }
    }
  }
}

// candidate epilogue: c = tanh(acc + b); h = u*h + (1-u)*c
__device__ __forceinline__ void write_cand(const v8f* acc, const float* bc,
    const float* sU, float* hbuf, int wave, int lane) {
  constexpr int TOTAL = MT * NT_C;                         // 12
  constexpr int TPW   = (TOTAL + NWAVES - 1) / NWAVES;     // 2
#pragma unroll
  for (int a = 0; a < TPW; ++a) {
    int ti = wave * TPW + a;
    if (ti < TOTAL) {
      int mt = ti / NT_C, nt = ti - mt * NT_C;
      int col   = nt * 16 + (lane & 15);
      int rbase = mt * 16 + ((lane >> 4) << 3);
#pragma unroll
      for (int r = 0; r < 8; ++r) {
        int row = rbase + r;
        if (row < ROWS) {
          float cnd  = tanhf(acc[a][r] + bc[col]);
          float u    = sU[row * HID + col];
          float hold = hbuf[row * HID + col];
          hbuf[row * HID + col] = u * hold + (1.f - u) * cnd;
        }
      }
    }
  }
}

__global__ __launch_bounds__(NTHREADS)
void dcrnn_main(const float* __restrict__ x, const float* __restrict__ supports,
                const __bf16* __restrict__ wfrag,
                const float* __restrict__ bg0, const float* __restrict__ bc0,
                const float* __restrict__ bg1, const float* __restrict__ bc1,
                const float* __restrict__ fcw, const float* __restrict__ fcb,
                float* __restrict__ out) {
  __shared__ float  sS[2 * NNODE * NNODE];
  __shared__ float  sHa[ROWS * HID];
  __shared__ float  sHb[ROWS * HID];
  __shared__ float  sR [ROWS * HID];
  __shared__ float  sU [ROWS * HID];
  __shared__ float  sCat[ROWS * 128];
  __shared__ float  sB1 [ROWS * 128];
  __shared__ float  sB2 [ROWS * 128];
  __shared__ __bf16 sFrag[4 * MT * 32 * 16];
  __shared__ __bf16 sBW[BSTAGE_ELEMS];     // TDM-staged weight fragments (32KB)
  __shared__ float  sLog[ROWS];

  const int tid  = threadIdx.x;
  const int wave = tid >> 5;
  const int lane = tid & 31;

  for (int i = tid; i < 2 * NNODE * NNODE; i += NTHREADS) sS[i] = supports[i];
  for (int i = tid; i < ROWS * HID; i += NTHREADS) { sHa[i] = 0.f; sHb[i] = 0.f; }
  __syncthreads();

  const int bBase = blockIdx.x * BB;
  v8f acc[MAXACC];

  for (int t = 0; t < TSTEPS; ++t) {
    const float* xg = x + ((size_t)bBase * TSTEPS + t) * (NNODE * DIMX);

    // ---- layer 0 cell (input x_t, state h_a) ----
    dconv<0, 96, NT_G>(xg, sS, sHa, sHb, sR, sCat, sB1, sB2, sFrag, sBW,
                       wfrag + OFF_WG0, acc, wave, lane);
    write_gate(acc, bg0, sR, sU, wave, lane);
    __syncthreads();
    dconv<1, 96, NT_C>(xg, sS, sHa, sHb, sR, sCat, sB1, sB2, sFrag, sBW,
                       wfrag + OFF_WC0, acc, wave, lane);
    write_cand(acc, bc0, sU, sHa, wave, lane);
    __syncthreads();

    // ---- layer 1 cell (input h_a, state h_b) ----
    dconv<2, 128, NT_G>(nullptr, sS, sHa, sHb, sR, sCat, sB1, sB2, sFrag, sBW,
                        wfrag + OFF_WG1, acc, wave, lane);
    write_gate(acc, bg1, sR, sU, wave, lane);
    __syncthreads();
    dconv<3, 128, NT_C>(nullptr, sS, sHa, sHb, sR, sCat, sB1, sB2, sFrag, sBW,
                        wfrag + OFF_WC1, acc, wave, lane);
    write_cand(acc, bc1, sU, sHb, wave, lane);
    __syncthreads();
  }

  // head: relu(h_b) @ fc_w + fc_b, then max over nodes per batch
  if (tid < ROWS) {
    float a = fcb[0];
#pragma unroll
    for (int c = 0; c < HID; ++c) a += fmaxf(sHb[tid * HID + c], 0.f) * fcw[c];
    sLog[tid] = a;
  }
  __syncthreads();
  if (tid < BB) {
    float m = -3.4e38f;
#pragma unroll
    for (int n = 0; n < NNODE; ++n) m = fmaxf(m, sLog[tid * NNODE + n]);
    out[bBase + tid] = m;
  }
}

extern "C" void kernel_launch(void* const* d_in, const int* in_sizes, int n_in,
                              void* d_out, int out_size, void* d_ws, size_t ws_size,
                              hipStream_t stream) {
  const float* x    = (const float*)d_in[0];
  const float* sup  = (const float*)d_in[1];
  const float* wg0  = (const float*)d_in[2];
  const float* bg0  = (const float*)d_in[3];
  const float* wc0  = (const float*)d_in[4];
  const float* bc0  = (const float*)d_in[5];
  const float* wg1  = (const float*)d_in[6];
  const float* bg1  = (const float*)d_in[7];
  const float* wc1  = (const float*)d_in[8];
  const float* bc1  = (const float*)d_in[9];
  const float* fcw  = (const float*)d_in[10];
  const float* fcb  = (const float*)d_in[11];
  __bf16* wsb = (__bf16*)d_ws;

  // pre-swizzle weights into WMMA B-fragment layout (bf16) in workspace
  dcrnn_prep_weights<<<(KT0*NT_G*512 + 255) / 256, 256, 0, stream>>>(wg0, wsb + OFF_WG0, 480, 128);
  dcrnn_prep_weights<<<(KT0*NT_C*512 + 255) / 256, 256, 0, stream>>>(wc0, wsb + OFF_WC0, 480, 64);
  dcrnn_prep_weights<<<(KT1*NT_G*512 + 255) / 256, 256, 0, stream>>>(wg1, wsb + OFF_WG1, 640, 128);
  dcrnn_prep_weights<<<(KT1*NT_C*512 + 255) / 256, 256, 0, stream>>>(wc1, wsb + OFF_WC1, 640, 64);

  dcrnn_main<<<BTOT / BB, NTHREADS, 0, stream>>>(
      x, sup, wsb, bg0, bc0, bg1, bc1, fcw, fcb, (float*)d_out);
}
